// CascadeQTAttB_21620865368156
// MI455X (gfx1250) — matrix operand order repaired and checked by reading, not compiled, not run on hardware
//
#include <hip/hip_runtime.h>

typedef __attribute__((ext_vector_type(16))) _Float16 v16h;
typedef __attribute__((ext_vector_type(8)))  float    v8f;

// Fixed problem shape from the reference harness.
constexpr int BS = 2, Cc = 256, H = 128, W = 128, NH = 8, DH = 32, KTOP = 8, K4 = 32;
constexpr int HB = H / 2, WB = W / 2, Lb = HB * WB, HW = H * W;

// A-matrix (16x32 f16) per-lane K mapping (ISA 7.12.2):
// lanes 0-15: V0..3 -> K 0..7, V4..7 -> K 16..23 ; lanes 16-31: +8
__device__ __forceinline__ int akmap(int e, int ksel) {
    return (e < 8) ? (ksel * 8 + e) : (16 + ksel * 8 + (e - 8));
}
// B-matrix (32x16 f16): same K mapping as A (lanes 0-15: K 0..7,16..23; lanes 16-31: +8)
__device__ __forceinline__ int bkmap(int e, int ksel) {
    return (e < 8) ? (ksel * 8 + e) : (16 + ksel * 8 + (e - 8));
}

__global__ __launch_bounds__(256) void qtatt_wmma_kernel(
    const float* __restrict__ qry, const float* __restrict__ key,
    const float* __restrict__ val, const long long* __restrict__ topk,
    const float* __restrict__ rel, float* __restrict__ msg,
    float* __restrict__ upidx)
{
    __shared__ float s_sc[NH][16 * 32];   // raw QK scores, C-layout dump
    __shared__ float s_pr[NH][16 * 32];   // softmax probabilities
    __shared__ int   s_ix[NH][32];        // gather indices (per candidate)

    const int blk  = blockIdx.x;
    const int b    = blk / Lb;
    const int l    = blk % Lb;
    const int by   = l / WB, bx = l % WB;
    const int wave = threadIdx.x >> 5;    // == head
    const int t    = threadIdx.x & 31;
    const int head = wave;
    const int m    = t & 15;              // M-row (A/C) or N-col (B/C)
    const int ksel = t >> 4;

    // ---- Phase 1: gather indices (lane t -> candidate t), zero pad rows ----
    {
        const int kk = t >> 2, o = t & 3;
        const long long base = (((long long)b * Lb + l) * KTOP + kk) * 2;
        const int r0 = (int)topk[base + 0];
        const int c0 = (int)topk[base + 1];
        int row = r0 * 2 + (o >> 1);   // offs: (0,0),(0,1),(1,0),(1,1)
        int col = c0 * 2 + (o & 1);
        int id  = row * W + col;
        id = id < 0 ? 0 : (id > HW - 1 ? HW - 1 : id);
        s_ix[wave][t] = id;
    }
#pragma unroll
    for (int r = 0; r < 12; ++r)      // prob rows 4..15 := 0 (dead M rows)
        s_pr[wave][(4 + r) * 32 + t] = 0.0f;
    __syncthreads();

    const float* qb = qry + ((long long)b * Cc + head * DH) * HW;
    const float* kb = key + ((long long)b * Cc + head * DH) * HW;
    const float* vb = val + ((long long)b * Cc + head * DH) * HW;

    // ---- Phase 2: QK^T via WMMA (scale folded into Q conversion) ----
    const float scale = 0.17677669529663687f;  // 1/sqrt(32)
    union { v16h v; _Float16 h[16]; } Aq, Bk0, Bk1;
    if (m < 4) {
        const int row = by * 2 + (m >> 1), col = bx * 2 + (m & 1);
        const int qoff = row * W + col;
#pragma unroll
        for (int e = 0; e < 16; ++e) {
            const int d = akmap(e, ksel);
            Aq.h[e] = (_Float16)(qb[d * HW + qoff] * scale);
        }
    } else {
#pragma unroll
        for (int e = 0; e < 16; ++e) Aq.h[e] = (_Float16)0.0f;
    }
    {
        const int i0 = s_ix[wave][m];        // candidate n = m        (tile 0)
        const int i1 = s_ix[wave][16 + m];   // candidate n = 16 + m   (tile 1)
#pragma unroll
        for (int e = 0; e < 16; ++e) {
            const int d = bkmap(e, ksel);    // channels strided by HW: scalar
            Bk0.h[e] = (_Float16)kb[d * HW + i0];
            Bk1.h[e] = (_Float16)kb[d * HW + i1];
        }
    }
    v8f c0 = {}, c1 = {};
    c0 = __builtin_amdgcn_wmma_f32_16x16x32_f16(false, Aq.v, false, Bk0.v,
                                                (short)0, c0, false, false);
    c1 = __builtin_amdgcn_wmma_f32_16x16x32_f16(false, Aq.v, false, Bk1.v,
                                                (short)0, c1, false, false);
#pragma unroll
    for (int j = 0; j < 8; ++j) {            // dump C-layout to LDS
        const int mr = ksel * 8 + j;
        s_sc[wave][mr * 32 + m]      = c0[j];
        s_sc[wave][mr * 32 + 16 + m] = c1[j];
    }
    __syncthreads();

    // ---- Phase 3: softmax over 32 candidates (+rel_pos) ----
    if (t < 4) {
        const int mr  = t;
        const int row = by * 2 + (mr >> 1), col = bx * 2 + (mr & 1);
        const float* rp =
            rel + ((((long long)b * NH + head) * H + row) * W + col) * K4;
        float s[32], mx = -3.4e38f;
#pragma unroll
        for (int n = 0; n < 32; ++n) {
            const float x = s_sc[wave][mr * 32 + n] + rp[n];
            s[n] = x;
            mx = fmaxf(mx, x);
        }
        float sum = 0.0f;
#pragma unroll
        for (int n = 0; n < 32; ++n) {
            const float e = __expf(s[n] - mx);
            s[n] = e;
            sum += e;
        }
        const float inv = 1.0f / sum;
#pragma unroll
        for (int n = 0; n < 32; ++n) s_pr[wave][mr * 32 + n] = s[n] * inv;
    }
    __syncthreads();

    // ---- Phase 4: P @ V via WMMA ----
    union { v16h v; _Float16 h[16]; } Pp, Bv0, Bv1;
#pragma unroll
    for (int e = 0; e < 16; ++e) {           // A = probs, K-dim = candidate n
        const int n = akmap(e, ksel);
        Pp.h[e] = (_Float16)s_pr[wave][m * 32 + n];
    }
    // V gather: candidate pairs (even n, n+1) are spatially adjacent
    // (idx[n+1] == idx[n]+1 by construction, 8B-aligned since col is even),
    // so load them as float2 -> global_load_b64, halving gather VMEM ops.
#pragma unroll
    for (int e = 0; e < 16; e += 2) {
        const int n  = bkmap(e, ksel);       // even candidate index
        const int id = s_ix[wave][n];
        const float2 p0 = *(const float2*)(vb + m * HW + id);         // dv = m
        const float2 p1 = *(const float2*)(vb + (16 + m) * HW + id);  // dv = 16+m
        Bv0.h[e]     = (_Float16)p0.x;
        Bv0.h[e + 1] = (_Float16)p0.y;
        Bv1.h[e]     = (_Float16)p1.x;
        Bv1.h[e + 1] = (_Float16)p1.y;
    }
    v8f o0 = {}, o1 = {};
    o0 = __builtin_amdgcn_wmma_f32_16x16x32_f16(false, Pp.v, false, Bv0.v,
                                                (short)0, o0, false, false);
    o1 = __builtin_amdgcn_wmma_f32_16x16x32_f16(false, Pp.v, false, Bv1.v,
                                                (short)0, o1, false, false);

    // ---- Phase 5: store msg (rows 0..3 live in lanes 0..15, j = m) ----
    if (ksel == 0) {
#pragma unroll
        for (int j = 0; j < 4; ++j) {
            const int row = by * 2 + (j >> 1), col = bx * 2 + (j & 1);
            const long long base =
                ((long long)b * HW + row * W + col) * Cc + head * DH;
            msg[base + m]      = o0[j];
            msg[base + 16 + m] = o1[j];
        }
    }
    // up_idx: only head-0 waves write (identical across heads)
    if (head == 0) {
        const float fid = (float)s_ix[wave][t];
#pragma unroll
        for (int j = 0; j < 4; ++j) {
            const int row = by * 2 + (j >> 1), col = bx * 2 + (j & 1);
            const long long ub = ((long long)b * HW + row * W + col) * K4;
            upidx[ub + t] = fid;
        }
    }
}

extern "C" void kernel_launch(void* const* d_in, const int* in_sizes, int n_in,
                              void* d_out, int out_size, void* d_ws, size_t ws_size,
                              hipStream_t stream) {
    const float*     qry  = (const float*)d_in[0];
    const float*     key  = (const float*)d_in[1];
    const float*     val  = (const float*)d_in[2];
    const long long* topk = (const long long*)d_in[3];
    const float*     rel  = (const float*)d_in[4];

    float* msg = (float*)d_out;
    float* up  = msg + (long long)BS * HW * Cc;   // second tuple output

    dim3 grid(BS * Lb), block(256);
    qtatt_wmma_kernel<<<grid, block, 0, stream>>>(qry, key, val, topk, rel,
                                                  msg, up);
}
